// HeteroEdgeEncoder_57243324121555
// MI455X (gfx1250) — matrix operand order repaired and checked by reading, not compiled, not run on hardware
//
#include <hip/hip_runtime.h>
#include <hip/hip_bf16.h>

// H=128 hidden, 3-layer edge MLP with LayerNorm+ReLU, bf16 WMMA compute,
// f32 accumulate. One launch per edge type.
//
// v3: epilogue LN param reads vectorized to f32x4 LDS loads (ds_load_b128)
//     instead of 96 scalar b32 loads per thread per layer.

#define HD   128       // hidden dim
#define MT   64        // edges (rows) per block
#define LDA  264       // activation LDS row stride (bf16 elems), padded vs bank conflicts
#define LDW  264       // transposed-weight LDS row stride (bf16 elems)
#define LDH  132       // f32 staging LDS row stride
#define NTHR 256       // 8 waves of 32

typedef __attribute__((ext_vector_type(16))) __bf16 v16bf;
typedef __attribute__((ext_vector_type(8)))  float  v8f;
typedef __attribute__((ext_vector_type(8)))  unsigned short u16x8;
typedef __attribute__((ext_vector_type(4)))  unsigned short u16x4;
typedef __attribute__((ext_vector_type(4)))  float  f32x4;

union FragBF { v16bf v; u16x8 h[2]; };

__device__ __forceinline__ unsigned short f2bf(float f) {
  // round-to-nearest-even f32 -> bf16
  unsigned int u = __float_as_uint(f);
  u += 0x7FFFu + ((u >> 16) & 1u);
  return (unsigned short)(u >> 16);
}

// Stage W (row-major [K][128] f32 in global) into LDS as bf16 W^T [128][K].
__device__ __forceinline__ void stage_weights(const float* __restrict__ W, int K,
                                              unsigned short* __restrict__ wt, int tid) {
  const int total = K * HD;
  for (int idx = tid; idx < total; idx += NTHR) {
    const int k = idx >> 7;          // row of W (coalesced global reads)
    const int n = idx & (HD - 1);    // col of W
    wt[n * LDW + k] = f2bf(W[idx]);
  }
}

// Stage per-layer LN params into LDS: pb[0..127]=bias, [128..255]=gamma, [256..383]=beta.
__device__ __forceinline__ void stage_params(const float* __restrict__ bl,
                                             const float* __restrict__ gl,
                                             const float* __restrict__ bt,
                                             float* __restrict__ pb, int tid) {
  for (int i = tid; i < 3 * HD; i += NTHR) {
    const float v = (i < HD) ? bl[i] : (i < 2 * HD) ? gl[i - HD] : bt[i - 2 * HD];
    pb[i] = v;
  }
}

// out[64,128] = act[64,K] @ W[K,128] via v_wmma_f32_16x16x32_bf16.
// Wave `wave` owns output columns [16*wave, 16*wave+16).
// B fragments are loaded once into registers and reused across all 4 row tiles.
template <int K>
__device__ __forceinline__ void layer_matmul(const unsigned short* __restrict__ act,
                                             const unsigned short* __restrict__ wt,
                                             float* __restrict__ hbuf,
                                             int lane, int wave) {
  const int m  = lane & 15;
  const int hi = lane >> 4;
  const int n0 = wave * 16;
  constexpr int NK = K / 32;

  // B 32x16 per k-chunk: lane holds column n0+m; lanes 0-15 K=k0..k0+15,
  // lanes 16-31 K=k0+16..k0+31. W^T rows contiguous in K -> two b128 LDS loads.
  FragBF b[NK];
  #pragma unroll
  for (int kk = 0; kk < NK; ++kk) {
    const unsigned short* bb = wt + (n0 + m) * LDW + kk * 32 + hi * 16;
    b[kk].h[0] = *(const u16x8*)(bb);
    b[kk].h[1] = *(const u16x8*)(bb + 8);
  }

  #pragma unroll
  for (int t = 0; t < MT / 16; ++t) {
    v8f acc = {0.f, 0.f, 0.f, 0.f, 0.f, 0.f, 0.f, 0.f};
    const int row0 = t * 16;
    #pragma unroll
    for (int kk = 0; kk < NK; ++kk) {
      // 16-bit A 16x32 layout: lane m holds row m; halves 0-15 = K k0+hi*16..+16
      // (contiguous, matching B)
      FragBF a;
      const unsigned short* ab = act + (row0 + m) * LDA + kk * 32;
      a.h[0] = *(const u16x8*)(ab + hi * 16);
      a.h[1] = *(const u16x8*)(ab + hi * 16 + 8);
      acc = __builtin_amdgcn_wmma_f32_16x16x32_bf16(false, a.v, false, b[kk].v,
                                                    (short)0, acc, false, false);
    }
    // C/D layout: VGPR j -> M = j + 8*hi, N = lane&15
    float* hr = hbuf + (row0 + hi * 8) * LDH + n0 + m;
    #pragma unroll
    for (int j = 0; j < 8; ++j) hr[j * LDH] = acc[j];
  }
}

// bias + LayerNorm + ReLU over each row of hbuf[64][128]; params come from LDS
// (vectorized f32x4 reads). Thread handles row tid>>2, 32 columns; row stats
// reduced over 4 consecutive lanes.
template <bool LAST>
__device__ __forceinline__ void ln_relu(const float* __restrict__ hbuf,
                                        unsigned short* __restrict__ act,
                                        const float* __restrict__ pb,   // LDS params
                                        float* __restrict__ outp,
                                        int e0, int nE, int tid) {
  const int r  = tid >> 2;
  const int c0 = (tid & 3) * 32;
  const float* hr    = hbuf + r * LDH + c0;
  const float* bl    = pb + c0;
  const float* gl    = pb + HD + c0;
  const float* betal = pb + 2 * HD + c0;
  float vals[32];
  float s = 0.f, s2 = 0.f;
  #pragma unroll
  for (int i = 0; i < 8; ++i) {
    f32x4 v  = *(const f32x4*)(hr + 4 * i);
    f32x4 bv = *(const f32x4*)(bl + 4 * i);
    #pragma unroll
    for (int j = 0; j < 4; ++j) {
      const float x = v[j] + bv[j];
      vals[4 * i + j] = x;
      s += x; s2 += x * x;
    }
  }
  // rows occupy lanes 4r..4r+3 inside a wave32 -> xor-reduce over masks 1,2
  s  += __shfl_xor(s, 1, 32);   s  += __shfl_xor(s, 2, 32);
  s2 += __shfl_xor(s2, 1, 32);  s2 += __shfl_xor(s2, 2, 32);
  const float mean = s * (1.f / 128.f);
  const float var  = s2 * (1.f / 128.f) - mean * mean;
  const float inv  = rsqrtf(var + 1e-5f);
  #pragma unroll
  for (int i = 0; i < 8; ++i) {
    f32x4 gv = *(const f32x4*)(gl + 4 * i);
    f32x4 tv = *(const f32x4*)(betal + 4 * i);
    f32x4 y;
    #pragma unroll
    for (int j = 0; j < 4; ++j) {
      float t = (vals[4 * i + j] - mean) * inv * gv[j] + tv[j];
      y[j] = fmaxf(t, 0.f);
    }
    if (!LAST) {
      u16x4 p;
      #pragma unroll
      for (int j = 0; j < 4; ++j) p[j] = f2bf(y[j]);
      *(u16x4*)(act + r * LDA + c0 + 4 * i) = p;
    } else {
      if (e0 + r < nE)
        *(f32x4*)(outp + (size_t)(e0 + r) * HD + c0 + 4 * i) = y;
    }
  }
}

__global__ __launch_bounds__(NTHR)
void hetero_edge_mlp(const float* __restrict__ xs, const float* __restrict__ xd,
                     const long long* __restrict__ ei,
                     const float* __restrict__ W0, const float* __restrict__ Ws,
                     const float* __restrict__ bias, const float* __restrict__ gam,
                     const float* __restrict__ betap,
                     float* __restrict__ outp, int nE) {
  extern __shared__ unsigned char smem[];
  unsigned short* act = (unsigned short*)smem;      // [MT][LDA] bf16 activations
  unsigned short* wt  = act + MT * LDA;             // [HD][LDW] bf16 W^T (per layer)
  float* hbuf  = (float*)(wt + HD * LDW);           // [MT][LDH] f32 staging
  float* pbuf0 = hbuf + MT * LDH;                   // [384] LN params, buffer 0
  float* pbuf1 = pbuf0 + 3 * HD;                    // [384] LN params, buffer 1

  const int tid  = threadIdx.x;
  const int lane = tid & 31;
  const int wave = tid >> 5;
  const int e0   = blockIdx.x * MT;

  // ---- gather: concat(x_src, x_dst) -> bf16 LDS (L2-resident node tables) ----
  {
    const int e   = tid >> 2;        // 0..63
    const int grp = tid & 3;         // 4 threads/edge, 64 cols each
    int egl = e0 + e; if (egl >= nE) egl = nE - 1;
    const long long src = ei[egl];
    const long long dst = ei[(long long)nE + egl];
    const float* xp = (grp < 2) ? (xs + src * HD) : (xd + dst * HD);
    const int cbase = (grp & 1) * 64;
    unsigned short* ar = act + e * LDA + grp * 64;
    #pragma unroll
    for (int i = 0; i < 16; ++i) {
      f32x4 v = *(const f32x4*)(xp + cbase + 4 * i);
      u16x4 p;
      #pragma unroll
      for (int j = 0; j < 4; ++j) p[j] = f2bf(v[j]);
      *(u16x4*)(ar + 4 * i) = p;
    }
  }
  stage_weights(W0, 2 * HD, wt, tid);
  stage_params(bias, gam, betap, pbuf0, tid);
  __syncthreads();

  // ---- layer 0: [64,256] @ [256,128] ----
  layer_matmul<2 * HD>(act, wt, hbuf, lane, wave);
  __syncthreads();
  ln_relu<false>(hbuf, act, pbuf0, outp, e0, nE, tid);
  stage_weights(Ws, HD, wt, tid);                                   // Ws[0]
  stage_params(bias + HD, gam + HD, betap + HD, pbuf1, tid);
  __syncthreads();

  // ---- layer 1: [64,128] @ [128,128] ----
  layer_matmul<HD>(act, wt, hbuf, lane, wave);
  __syncthreads();
  ln_relu<false>(hbuf, act, pbuf1, outp, e0, nE, tid);
  stage_weights(Ws + HD * HD, HD, wt, tid);                         // Ws[1]
  stage_params(bias + 2 * HD, gam + 2 * HD, betap + 2 * HD, pbuf0, tid);
  __syncthreads();

  // ---- layer 2: [64,128] @ [128,128] -> global f32 ----
  layer_matmul<HD>(act, wt, hbuf, lane, wave);
  __syncthreads();
  ln_relu<true>(hbuf, act, pbuf0, outp, e0, nE, tid);
}

extern "C" void kernel_launch(void* const* d_in, const int* in_sizes, int n_in,
                              void* d_out, int out_size, void* d_ws, size_t ws_size,
                              hipStream_t stream) {
  (void)n_in; (void)out_size; (void)d_ws; (void)ws_size;
  const float*     x_b   = (const float*)d_in[0];
  const float*     x_e   = (const float*)d_in[1];
  const long long* ei_bb = (const long long*)d_in[2];
  const long long* ei_be = (const long long*)d_in[3];
  const long long* ei_ee = (const long long*)d_in[4];
  const int E = in_sizes[2] / 2;           // ei has shape (2, E)
  float* out = (float*)d_out;
  const int blocks = (E + MT - 1) / MT;
  const size_t shmem = (size_t)(MT * LDA + HD * LDW) * sizeof(unsigned short)
                     + (size_t)(MT * LDH + 6 * HD) * sizeof(float);  // ~138 KB / block

  // barrel -> barrel
  hetero_edge_mlp<<<blocks, NTHR, shmem, stream>>>(
      x_b, x_b, ei_bb,
      (const float*)d_in[5], (const float*)d_in[6], (const float*)d_in[7],
      (const float*)d_in[8], (const float*)d_in[9],
      out, E);
  // barrel -> endcap
  hetero_edge_mlp<<<blocks, NTHR, shmem, stream>>>(
      x_b, x_e, ei_be,
      (const float*)d_in[10], (const float*)d_in[11], (const float*)d_in[12],
      (const float*)d_in[13], (const float*)d_in[14],
      out + (size_t)E * HD, E);
  // endcap -> endcap
  hetero_edge_mlp<<<blocks, NTHR, shmem, stream>>>(
      x_e, x_e, ei_ee,
      (const float*)d_in[15], (const float*)d_in[16], (const float*)d_in[17],
      (const float*)d_in[18], (const float*)d_in[19],
      out + 2 * (size_t)E * HD, E);
}
